// SNN_Only_52355651338861
// MI455X (gfx1250) — matrix operand order repaired and checked by reading, not compiled, not run on hardware
//
#include <hip/hip_runtime.h>
#include <hip/hip_fp16.h>

typedef __attribute__((ext_vector_type(16))) _Float16 v16h;
typedef __attribute__((ext_vector_type(8)))  _Float16 v8h;
typedef __attribute__((ext_vector_type(8)))  float    v8f;

static constexpr int B_  = 4096;
static constexpr int D_  = 1024;
static constexpr int H1_ = 2048;
static constexpr int H2_ = 1024;
static constexpr int C_  = 64;
static constexpr int T_  = 20;
static constexpr float BETA = 0.9f;
static constexpr float THR  = 1.0f;

// The async-to-LDS builtin being declared is our proxy for the gfx1250 async
// instructions existing in this toolchain's backend; we emit them via inline
// asm (portable across builtin-signature differences).
#if defined(__gfx1250__) && __has_builtin(__builtin_amdgcn_global_load_async_to_lds_b128)
#define USE_ASYNC_LDS 1
#else
#define USE_ASYNC_LDS 0
#endif

#define LDS_AS __attribute__((address_space(3)))

__device__ inline void wait_async0() {
#if USE_ASYNC_LDS
#if __has_builtin(__builtin_amdgcn_s_wait_asynccnt)
  __builtin_amdgcn_s_wait_asynccnt(0);
#else
  asm volatile("s_wait_asynccnt 0x0" ::: "memory");
#endif
#endif
}

#if USE_ASYNC_LDS
// 16B global -> LDS direct copy via the CDNA5 async unit (ASYNCcnt-tracked).
// VDST = LDS byte address VGPR, VADDR = 64-bit global address, no SADDR.
__device__ inline void async_cp16(const _Float16* g, _Float16* l) {
  const unsigned lds = (unsigned)(size_t)(LDS_AS void*)l;
  asm volatile("global_load_async_to_lds_b128 %0, %1, off"
               :: "v"(lds), "v"(g) : "memory");
}
#endif

// ---------------------------------------------------------------------------
// Elementwise f32 -> f16 convert (optionally scaled).
// ---------------------------------------------------------------------------
__global__ void cvt_scale_f16(const float* __restrict__ src,
                              _Float16* __restrict__ dst, int n, float scale) {
  int i = blockIdx.x * blockDim.x + threadIdx.x;
  const int st = gridDim.x * blockDim.x;
  for (; i < n; i += st) dst[i] = (_Float16)(src[i] * scale);
}

__device__ inline v16h cat16(v8h lo, v8h hi) {
  v16h r;
#pragma unroll
  for (int i = 0; i < 8; ++i) { r[i] = lo[i]; r[i + 8] = hi[i]; }
  return r;
}

// ---------------------------------------------------------------------------
// Fused GEMM (H = A @ W^T + bias) + LIF epilogue.
// f16 inputs, f32 accumulate via v_wmma_f32_16x16x32_f16.
// LDS ping-pong with async global->LDS staging (GLOBAL_LOAD_ASYNC_TO_LDS_B128)
// when available; register double-buffering fallback otherwise.
// MODE 0: LIF (mem, spk).  MODE 1: LIF + spike-sum.  MODE 2: plain h -> aux.
// ---------------------------------------------------------------------------
template <int BM, int BN, int WM, int WN, int WAVES_M, int WAVES_N, int MODE>
__global__ __launch_bounds__(WAVES_M * WAVES_N * 32)
void gemm_wmma_lif(const _Float16* __restrict__ A,   // [M,K]
                   const _Float16* __restrict__ W,   // [N,K]
                   const float* __restrict__ bias,   // [N]
                   float* __restrict__ mem,          // [M,N]
                   _Float16* __restrict__ spk,       // [M,N]
                   float* __restrict__ aux,          // ssum or out [M,N]
                   int M, int N, int K) {
  constexpr int THREADS = WAVES_M * WAVES_N * 32;
  constexpr int LDT = 40;            // halves/row: 80B pitch, bank-friendly
  constexpr int ACPT = (BM * 4) / THREADS;  // 16B chunks per thread (A tile)
  constexpr int BCPT = (BN * 4) / THREADS;  // 16B chunks per thread (B tile)
  __shared__ __align__(16) _Float16 As[2][BM * LDT];
  __shared__ __align__(16) _Float16 Bs[2][BN * LDT];

  const int tid  = threadIdx.x;
  const int lane = tid & 31;
  const int wid  = tid >> 5;
  const int wm   = wid % WAVES_M;
  const int wn   = wid / WAVES_M;
  const int g    = lane >> 4;   // half-wave group (K sub-range select)
  const int ln   = lane & 15;   // row (A) / column (B, D) within 16x16 tile

  const int mblk = blockIdx.y * BM;
  const int nblk = blockIdx.x * BN;

  v8f acc[WM][WN];
  {
    v8f z = {};
#pragma unroll
    for (int i = 0; i < WM; ++i)
#pragma unroll
      for (int j = 0; j < WN; ++j) acc[i][j] = z;
  }

#if USE_ASYNC_LDS
  // Direct DMA staging: no data VGPRs, tracked with ASYNCcnt.
  auto stage_tile = [&](int kb, int buf) {
#pragma unroll
    for (int u = 0; u < ACPT; ++u) {
      const int c = tid + u * THREADS, row = c >> 2, col = (c & 3) << 3;
      async_cp16(&A[(size_t)(mblk + row) * K + kb + col],
                 &As[buf][row * LDT + col]);
    }
#pragma unroll
    for (int u = 0; u < BCPT; ++u) {
      const int c = tid + u * THREADS, row = c >> 2, col = (c & 3) << 3;
      async_cp16(&W[(size_t)(nblk + row) * K + kb + col],
                 &Bs[buf][row * LDT + col]);
    }
  };
#else
  v8h areg[ACPT], breg[BCPT];
  auto load_tile = [&](int kb) {
#pragma unroll
    for (int u = 0; u < ACPT; ++u) {
      const int c = tid + u * THREADS, row = c >> 2, col = (c & 3) << 3;
      areg[u] = *(const v8h*)&A[(size_t)(mblk + row) * K + kb + col];
    }
#pragma unroll
    for (int u = 0; u < BCPT; ++u) {
      const int c = tid + u * THREADS, row = c >> 2, col = (c & 3) << 3;
      breg[u] = *(const v8h*)&W[(size_t)(nblk + row) * K + kb + col];
    }
  };
  auto store_tile = [&](int buf) {
#pragma unroll
    for (int u = 0; u < ACPT; ++u) {
      const int c = tid + u * THREADS, row = c >> 2, col = (c & 3) << 3;
      *(v8h*)&As[buf][row * LDT + col] = areg[u];
    }
#pragma unroll
    for (int u = 0; u < BCPT; ++u) {
      const int c = tid + u * THREADS, row = c >> 2, col = (c & 3) << 3;
      *(v8h*)&Bs[buf][row * LDT + col] = breg[u];
    }
  };
#endif

  auto compute = [&](int buf) {
    v16h afr[WM], bfr[WN];
#pragma unroll
    for (int i = 0; i < WM; ++i) {
      // A 16x32 layout: half-wave g holds K = 8g..8g+7 and 16+8g..16+8g+7.
      const _Float16* base = &As[buf][(wm * WM * 16 + i * 16 + ln) * LDT];
      afr[i] = cat16(*(const v8h*)&base[8 * g], *(const v8h*)&base[16 + 8 * g]);
    }
#pragma unroll
    for (int j = 0; j < WN; ++j) {
      // B 32x16 layout: half-wave g holds K = 8g..8g+7 and 16+8g..16+8g+7 for column ln.
      const _Float16* base = &Bs[buf][(wn * WN * 16 + j * 16 + ln) * LDT];
      bfr[j] = cat16(*(const v8h*)&base[8 * g], *(const v8h*)&base[16 + 8 * g]);
    }
#pragma unroll
    for (int i = 0; i < WM; ++i)
#pragma unroll
      for (int j = 0; j < WN; ++j)
        acc[i][j] = __builtin_amdgcn_wmma_f32_16x16x32_f16(
            false, afr[i], false, bfr[j], (short)0, acc[i][j], false, false);
  };

  const int ktiles = K >> 5;  // BK == WMMA K == 32

#if USE_ASYNC_LDS
  stage_tile(0, 0);
  wait_async0();              // own async writes into LDS landed
  __syncthreads();            // => all waves' tile-0 data visible

  int buf = 0;
  for (int kt = 0; kt < ktiles - 1; ++kt) {
    stage_tile((kt + 1) << 5, buf ^ 1);  // DMA next tile during compute
    compute(buf);                        // ds_load fragments + 8 WMMAs
    wait_async0();                       // own DMA into buf^1 complete
    __syncthreads();                     // single barrier per K-tile
    buf ^= 1;
  }
  compute(buf);
#else
  load_tile(0);
  store_tile(0);
  __syncthreads();

  int buf = 0;
  for (int kt = 0; kt < ktiles - 1; ++kt) {
    load_tile((kt + 1) << 5);   // global loads in flight during compute
    compute(buf);
    store_tile(buf ^ 1);
    __syncthreads();
    buf ^= 1;
  }
  compute(buf);
#endif

  // Epilogue. D layout: VGPR r -> row (r + 8g), column ln.
#pragma unroll
  for (int i = 0; i < WM; ++i) {
    const int grow0 = mblk + wm * WM * 16 + i * 16 + 8 * g;
#pragma unroll
    for (int j = 0; j < WN; ++j) {
      const int gcol = nblk + wn * WN * 16 + j * 16 + ln;
      const float bv = bias[gcol];
#pragma unroll
      for (int r = 0; r < 8; ++r) {
        const size_t idx = (size_t)(grow0 + r) * N + gcol;
        const float h = acc[i][j][r] + bv;
        if (MODE == 2) {
          aux[idx] = h;
        } else {
          const float mo = mem[idx];                        // pre-update mem
          const float mn = BETA * mo + h - (mo > THR ? THR : 0.0f);
          mem[idx] = mn;
          const float s = (mn > THR) ? 1.0f : 0.0f;         // heaviside
          spk[idx] = (_Float16)s;
          if (MODE == 1) aux[idx] += s;
        }
      }
    }
  }
}

// ---------------------------------------------------------------------------
extern "C" void kernel_launch(void* const* d_in, const int* in_sizes, int n_in,
                              void* d_out, int out_size, void* d_ws,
                              size_t ws_size, hipStream_t stream) {
  (void)in_sizes; (void)n_in; (void)out_size; (void)ws_size;
  const float* x  = (const float*)d_in[0];
  const float* W1 = (const float*)d_in[1];
  const float* b1 = (const float*)d_in[2];
  const float* W2 = (const float*)d_in[3];
  const float* b2 = (const float*)d_in[4];
  const float* Wo = (const float*)d_in[5];
  const float* bo = (const float*)d_in[6];
  float* out = (float*)d_out;

  char* ws = (char*)d_ws;
  size_t off = 0;
  auto carve = [&](size_t bytes) -> char* {
    char* p = ws + off;
    off = (off + bytes + 255) & ~(size_t)255;
    return p;
  };
  _Float16* w1h = (_Float16*)carve((size_t)H1_ * D_ * 2);
  _Float16* w2h = (_Float16*)carve((size_t)H2_ * H1_ * 2);
  _Float16* woh = (_Float16*)carve((size_t)C_ * H2_ * 2);
  _Float16* xh  = (_Float16*)carve((size_t)B_ * D_ * 2);
  _Float16* s1h = (_Float16*)carve((size_t)B_ * H1_ * 2);
  _Float16* s2h = (_Float16*)carve((size_t)B_ * H2_ * 2);
  _Float16* ssh = (_Float16*)carve((size_t)B_ * H2_ * 2);
  float* mem1 = (float*)carve((size_t)B_ * H1_ * 4);
  float* mem2 = (float*)carve((size_t)B_ * H2_ * 4);
  float* ssum = (float*)carve((size_t)B_ * H2_ * 4);

  // Persistent state starts at zero every call (graph-capture safe).
  (void)hipMemsetAsync(mem1, 0, (size_t)B_ * H1_ * 4, stream);
  (void)hipMemsetAsync(mem2, 0, (size_t)B_ * H2_ * 4, stream);
  (void)hipMemsetAsync(ssum, 0, (size_t)B_ * H2_ * 4, stream);

  auto cvt = [&](const float* s, _Float16* d, int n, float sc) {
    int blocks = (n + 256 * 8 - 1) / (256 * 8);
    cvt_scale_f16<<<blocks, 256, 0, stream>>>(s, d, n, sc);
  };
  cvt(W1, w1h, H1_ * D_, 1.0f);
  cvt(W2, w2h, H2_ * H1_, 1.0f);
  cvt(Wo, woh, C_ * H2_, 1.0f);
  cvt(x,  xh,  B_ * D_, 1.0f);

  const dim3 blk(256);
  const dim3 g1(H1_ / 128, B_ / 128);
  const dim3 g2(H2_ / 128, B_ / 128);
  for (int t = 0; t < T_; ++t) {
    const _Float16* cur = (t == 0) ? xh : s2h;  // D_ == H2_ == 1024
    gemm_wmma_lif<128, 128, 2, 4, 4, 2, 0><<<g1, blk, 0, stream>>>(
        cur, w1h, b1, mem1, s1h, nullptr, B_, H1_, D_);
    gemm_wmma_lif<128, 128, 2, 4, 4, 2, 1><<<g2, blk, 0, stream>>>(
        s1h, w2h, b2, mem2, s2h, ssum, B_, H2_, H1_);
  }

  // Readout: (spk_sum / T) @ Wo^T + bo
  cvt(ssum, ssh, B_ * H2_, 1.0f / (float)T_);
  const dim3 go(C_ / 64, B_ / 128);
  gemm_wmma_lif<128, 64, 2, 2, 4, 2, 2><<<go, blk, 0, stream>>>(
      ssh, woh, bo, nullptr, nullptr, out, B_, C_, H2_);
}
